// NoHubUniformLoss_40931038331407
// MI455X (gfx1250) — matrix operand order repaired and checked by reading, not compiled, not run on hardware
//
#include <hip/hip_runtime.h>
#include <hip/hip_bf16.h>
#include <math.h>

// ---------------------------------------------------------------------------
// NoHubUniformLoss on MI455X (gfx1250, wave32, WMMA)
//   logits[b,n,m] = 0.5 * <emb[b,n], emb[b,m]>,  loss = mean_b logsumexp(logits[b])
//   B=8, N=4096, D=64
// Symmetric-triangle version: exp(logits) is symmetric, so
//   S_b = sum(diag 128x128 blocks) + 2 * sum(strict-upper 128x128 blocks)
// -> only 528 of 1024 block-pairs computed (~1.94x less WMMA + exp work).
// Epilogue uses explicit exp trees + dual accumulators to avoid serial
// TRANS->ADD dependency chains (v_exp_f32 co-issues with WMMA/VALU).
// ---------------------------------------------------------------------------

typedef __attribute__((ext_vector_type(16))) _Float16 v16h;
typedef __attribute__((ext_vector_type(8)))  _Float16 v8h;
typedef __attribute__((ext_vector_type(4)))  _Float16 v4h;
typedef __attribute__((ext_vector_type(8)))  float    v8f;
typedef __attribute__((ext_vector_type(4)))  float    v4f;

#define NBATCH 8
#define NROWS  4096
#define NDIM   64
#define SLABS  32                          // 32 slabs of 128 rows per batch
#define NPAIRS (SLABS * (SLABS + 1) / 2)   // 528 triangular slab pairs

// ---------------------------------------------------------------------------
// Kernel 0: zero the per-batch max-norm bit slots (atomicMax targets).
// ---------------------------------------------------------------------------
__global__ void nhul_init_kernel(unsigned* __restrict__ mbits) {
    if (threadIdx.x < NBATCH) mbits[threadIdx.x] = 0u;
}

// ---------------------------------------------------------------------------
// Kernel 1: f32 -> f16 convert + per-row squared norms -> per-batch max.
// One thread per row (64 elements). 256 rows/block => whole block is one batch.
// atomicMax on float bits is valid for non-negative floats and is
// order-independent => deterministic.
// ---------------------------------------------------------------------------
__global__ __launch_bounds__(256)
void nhul_convert_kernel(const float* __restrict__ emb,
                         _Float16* __restrict__ embh,
                         unsigned* __restrict__ mbits) {
    const int row = blockIdx.x * 256 + threadIdx.x;          // 0 .. 32767
    const float*    src = emb  + (size_t)row * NDIM;
    _Float16*       dst = embh + (size_t)row * NDIM;

    float sq = 0.f;
#pragma unroll
    for (int i = 0; i < NDIM / 4; ++i) {
        v4f x = *(const v4f*)(src + 4 * i);
        sq += x.x * x.x + x.y * x.y + x.z * x.z + x.w * x.w;
        v4h h;
        h.x = (_Float16)x.x; h.y = (_Float16)x.y;
        h.z = (_Float16)x.z; h.w = (_Float16)x.w;
        *(v4h*)(dst + 4 * i) = h;
    }

    __shared__ float smax[256];
    smax[threadIdx.x] = sq;
    __syncthreads();
#pragma unroll
    for (int off = 128; off > 0; off >>= 1) {
        if (threadIdx.x < off)
            smax[threadIdx.x] = fmaxf(smax[threadIdx.x], smax[threadIdx.x + off]);
        __syncthreads();
    }
    if (threadIdx.x == 0) {
        const int b = blockIdx.x >> 4;                       // 16 blocks per batch
        atomicMax(mbits + b, __float_as_uint(smax[0]));
    }
}

// ---------------------------------------------------------------------------
// Per-tile epilogue: 8 independent v_exp_f32 then a pairwise add tree
// (explicit: default FP rules forbid reassociation, so spelling out the
// tree is what breaks the serial TRANS->ADD chain).
// ---------------------------------------------------------------------------
__device__ __forceinline__ float nhul_tile_expsum(const v8f c, float negMb) {
    float e0 = __expf(fmaf(0.5f, c[0], negMb));
    float e1 = __expf(fmaf(0.5f, c[1], negMb));
    float e2 = __expf(fmaf(0.5f, c[2], negMb));
    float e3 = __expf(fmaf(0.5f, c[3], negMb));
    float e4 = __expf(fmaf(0.5f, c[4], negMb));
    float e5 = __expf(fmaf(0.5f, c[5], negMb));
    float e6 = __expf(fmaf(0.5f, c[6], negMb));
    float e7 = __expf(fmaf(0.5f, c[7], negMb));
    return ((e0 + e1) + (e2 + e3)) + ((e4 + e5) + (e6 + e7));
}

// ---------------------------------------------------------------------------
// Kernel 2: fused Gram block (WMMA f16 -> f32) + exp-sum epilogue, upper
// triangle only. Grid: 8 batches * 528 slab-pairs. Block: 256 threads
// = 8 wave32s. Slab pair (i,j), i<=j: rows = slab i (128), cols = slab j
// (128). Each wave owns a fixed 16-row A tile (all D=64 in two v16h regs)
// and sweeps the pair's 8 column tiles (fully unrolled, two tiles in flight
// with independent accumulators; all B loads via immediate offsets from one
// per-lane base). Off-diagonal pairs weighted 2x (symmetry of exp(logits)).
// Max logit M_b = 0.5*max||e||^2 is exact (Cauchy-Schwarz, attained on the
// diagonal), so every exponent is <= 0: one expf per element, no running max.
// ---------------------------------------------------------------------------
__global__ __launch_bounds__(256)
void nhul_gram_lse_kernel(const _Float16* __restrict__ embh,
                          const unsigned* __restrict__ mbits,
                          float* __restrict__ parts) {
    const int wg   = blockIdx.x;
    const int b    = wg / NPAIRS;        // batch
    int       p    = wg - b * NPAIRS;    // triangular pair index

    // Decode p -> (i, j), i <= j (uniform scalar loop, <=32 iterations).
    int i = 0;
    while (p >= (SLABS - i)) { p -= (SLABS - i); ++i; }
    const int j = i + p;

    const int tid  = threadIdx.x;
    const int wave = tid >> 5;
    const int lane = tid & 31;
    const int m    = lane & 15;          // row within A tile / col within B tile
    const int h    = lane >> 4;          // lane half-group

    const _Float16* base = embh + (size_t)b * NROWS * NDIM;

    // ---- A tile (16 rows x 64 k), resident in registers for whole kernel ---
    // A layout (matches B): lane m=L%16 holds contiguous k in [16h, 16h+16).
    // a0 covers k 0..31, a1 k 32..63.
    const _Float16* aPtr = base + (size_t)(i * 128 + wave * 16 + m) * NDIM;
    union { v16h v; v8h pr[2]; } a0, a1;
    a0.pr[0] = *(const v8h*)(aPtr +      16 * h);
    a0.pr[1] = *(const v8h*)(aPtr +  8 + 16 * h);
    a1.pr[0] = *(const v8h*)(aPtr + 32 + 16 * h);
    a1.pr[1] = *(const v8h*)(aPtr + 40 + 16 * h);

    const float negMb = -0.5f * __uint_as_float(mbits[b]);
    float s0 = 0.f, s1 = 0.f;

    // ---- B tiles (64 k x 16 cols). B layout (same k order as A): lane n=L%16 holds
    // contiguous k in [16h, 16h+16) per v16h. Single per-lane base pointer;
    // 8 tiles in slab j at immediate offsets of t*1024 halves (2048 B).
    const _Float16* colBase = base + (size_t)(j * 128 + m) * NDIM + 16 * h;

#pragma unroll
    for (int u = 0; u < 4; ++u) {
        const int tA = 2 * u, tB = 2 * u + 1;

        // Speculative prefetch of the next tile pair (OOB-safe, dropped).
        __builtin_prefetch(colBase + (tB + 1) * (16 * NDIM), 0, 3);

        v16h bA0 = *(const v16h*)(colBase + tA * (16 * NDIM));        // k 0..31
        v16h bA1 = *(const v16h*)(colBase + tA * (16 * NDIM) + 32);   // k 32..63
        v16h bB0 = *(const v16h*)(colBase + tB * (16 * NDIM));
        v16h bB1 = *(const v16h*)(colBase + tB * (16 * NDIM) + 32);

        v8f cA = {}, cB = {};
        cA = __builtin_amdgcn_wmma_f32_16x16x32_f16(false, a0.v, false, bA0,
                                                    (short)0, cA, false, false);
        cA = __builtin_amdgcn_wmma_f32_16x16x32_f16(false, a1.v, false, bA1,
                                                    (short)0, cA, false, false);
        cB = __builtin_amdgcn_wmma_f32_16x16x32_f16(false, a0.v, false, bB0,
                                                    (short)0, cB, false, false);
        cB = __builtin_amdgcn_wmma_f32_16x16x32_f16(false, a1.v, false, bB1,
                                                    (short)0, cB, false, false);

        s0 += nhul_tile_expsum(cA, negMb);
        s1 += nhul_tile_expsum(cB, negMb);
    }

    float s = s0 + s1;
    // Off-diagonal block counts for both (i,j) and (j,i).
    if (i != j) s *= 2.0f;

    // ---- block reduction -> one deterministic partial per WG ----
    __shared__ float red[256];
    red[tid] = s;
    __syncthreads();
#pragma unroll
    for (int off = 128; off > 0; off >>= 1) {
        if (tid < off) red[tid] += red[tid + off];
        __syncthreads();
    }
    if (tid == 0) parts[wg] = red[0];
}

// ---------------------------------------------------------------------------
// Kernel 3: combine 528 partials per batch (fixed order => deterministic),
// loss_b = M_b + log(S_b); output = mean over batches.
// ---------------------------------------------------------------------------
__global__ void nhul_final_kernel(const unsigned* __restrict__ mbits,
                                  const float* __restrict__ parts,
                                  float* __restrict__ out) {
    __shared__ float loss[NBATCH];
    const int t = threadIdx.x;
    if (t < NBATCH) {
        float sb = 0.f;
        for (int k = 0; k < NPAIRS; ++k) sb += parts[t * NPAIRS + k];
        loss[t] = 0.5f * __uint_as_float(mbits[t]) + __logf(sb);
    }
    __syncthreads();
    if (t == 0) {
        float acc = 0.f;
#pragma unroll
        for (int bb = 0; bb < NBATCH; ++bb) acc += loss[bb];
        out[0] = acc * (1.0f / NBATCH);
    }
}

// ---------------------------------------------------------------------------
extern "C" void kernel_launch(void* const* d_in, const int* in_sizes, int n_in,
                              void* d_out, int out_size, void* d_ws, size_t ws_size,
                              hipStream_t stream) {
    (void)in_sizes; (void)n_in; (void)out_size; (void)ws_size;

    const float* emb = (const float*)d_in[0];
    float* out = (float*)d_out;

    // Workspace layout:
    //   [0, 4MB)        : f16 embeddings (8*4096*64 halves)
    //   [4MB, +32B)     : per-batch max-norm bits (8 x u32)
    //   next            : per-WG partial sums (8*528 floats)
    _Float16* embh  = (_Float16*)d_ws;
    unsigned* mbits = (unsigned*)((char*)d_ws +
                        (size_t)NBATCH * NROWS * NDIM * sizeof(_Float16));
    float*    parts = (float*)(mbits + NBATCH);

    nhul_init_kernel<<<1, 32, 0, stream>>>(mbits);
    nhul_convert_kernel<<<(NBATCH * NROWS) / 256, 256, 0, stream>>>(emb, embh, mbits);
    nhul_gram_lse_kernel<<<NBATCH * NPAIRS, 256, 0, stream>>>(embh, mbits, parts);
    nhul_final_kernel<<<1, 32, 0, stream>>>(mbits, parts, out);
}
